// PromptDualCLBERT_78984448574122
// MI455X (gfx1250) — matrix operand order repaired and checked by reading, not compiled, not run on hardware
//
#include <hip/hip_runtime.h>
#include <hip/hip_bf16.h>
#include <math.h>

// ---------------- problem constants ----------------
#define BN_   4096          // batch
#define HN_   768           // hidden
#define LN_   5             // labels
#define SMN_  20480         // BN_*LN_ rows of theta-flat
#define INV_T 14.2857142857142857f   // 1/0.07

typedef _Float16 h16 __attribute__((ext_vector_type(16)));
typedef _Float16 h8  __attribute__((ext_vector_type(8)));
typedef float    f8  __attribute__((ext_vector_type(8)));
typedef float    f2  __attribute__((ext_vector_type(2)));

// ---------------------------------------------------------------------------
// Kernel 0: zero the accumulators (pos_z/neg_z/pos_t/neg_t = 16384 floats,
// contiguous in ws) and the whole d_out (20481 floats, atomically accumulated).
// ---------------------------------------------------------------------------
__global__ void k_init(float* __restrict__ acc4, float* __restrict__ out) {
    int idx = blockIdx.x * 256 + threadIdx.x;
    if (idx < 4 * BN_)                 acc4[idx] = 0.0f;
    else if (idx < 4 * BN_ + SMN_ + 1) out[idx - 4 * BN_] = 0.0f;
}

// ---------------------------------------------------------------------------
// Kernel 1: extract z / theta rows, write f32 copies + f16 L2-normalized
// copies.  One block per batch element, 6 waves = 6 output rows.
// ---------------------------------------------------------------------------
__global__ void k_extract(const float* __restrict__ seq,
                          float* __restrict__ z, float* __restrict__ th,
                          _Float16* __restrict__ znh, _Float16* __restrict__ thh) {
    const int b    = blockIdx.x;
    const int w    = threadIdx.x >> 5;     // 0..5: 0 -> z, 1..5 -> theta rows
    const int lane = threadIdx.x & 31;
    const float* base = seq + (size_t)b * 64 * HN_;

    float v[24];
    float ss = 0.0f;
#pragma unroll
    for (int t = 0; t < 24; ++t) {
        int h = lane + 32 * t;
        float x;
        if (w == 0)      x = base[h];
        else if (w == 1) x = 0.5f * (base[HN_ + h] + base[2 * HN_ + h]);
        else             x = base[(w + 2) * HN_ + h];
        v[t] = x;
        ss += x * x;
    }
#pragma unroll
    for (int m = 1; m < 32; m <<= 1) ss += __shfl_xor(ss, m, 32);
    float rn = 1.0f / fmaxf(sqrtf(ss), 1e-12f);

    float*    drow;
    _Float16* hrow;
    if (w == 0) { drow = z  + (size_t)b * HN_;                hrow = znh + (size_t)b * HN_; }
    else        { size_t r = (size_t)b * LN_ + (w - 1);
                  drow = th + r * HN_;                        hrow = thh + r * HN_; }
#pragma unroll
    for (int t = 0; t < 24; ++t) {
        int h = lane + 32 * t;
        drow[h] = v[t];
        hrow[h] = (_Float16)(v[t] * rn);
    }
}

// ---------------------------------------------------------------------------
// Kernel 2: per-feature BN stats folded to affine:  bn(x) = x*s[h] + t[h].
// blocks 0..2 -> z columns (4096 rows), blocks 3..5 -> theta cols (20480 rows)
// ---------------------------------------------------------------------------
__global__ void k_colstats(const float* __restrict__ z, const float* __restrict__ th,
                           const float* __restrict__ cg, const float* __restrict__ cb,
                           const float* __restrict__ lg, const float* __restrict__ lb,
                           float* __restrict__ sz, float* __restrict__ tz,
                           float* __restrict__ st, float* __restrict__ tt) {
    int col = (blockIdx.x % 3) * 256 + threadIdx.x;   // 0..767
    if (blockIdx.x < 3) {
        double s = 0.0, q = 0.0;
        for (int r = 0; r < BN_; ++r) { float x = z[(size_t)r * HN_ + col]; s += x; q += (double)x * x; }
        float mean = (float)(s / BN_);
        float var  = (float)(q / BN_) - mean * mean;
        float sc   = cg[col] * rsqrtf(var + 1e-5f);
        sz[col] = sc;  tz[col] = cb[col] - mean * sc;
    } else {
        double s = 0.0, q = 0.0;
        for (int r = 0; r < SMN_; ++r) { float x = th[(size_t)r * HN_ + col]; s += x; q += (double)x * x; }
        float mean = (float)(s / SMN_);
        float var  = (float)(q / SMN_) - mean * mean;
        float sc   = lg[col] * rsqrtf(var + 1e-5f);
        st[col] = sc;  tt[col] = lb[col] - mean * sc;
    }
}

// ---------------------------------------------------------------------------
// fp32 WMMA helper: one 16x16 tile, K-loop over 768 with V_WMMA_F32_16X16X4_F32.
// A = bn(X) on the fly, B = W^T (lane reads its own W row contiguously).
// ---------------------------------------------------------------------------
__device__ __forceinline__ f8 gemm_f32_tile(const float* __restrict__ Xrow,
                                            const float* __restrict__ Wrow,
                                            const float* __restrict__ s,
                                            const float* __restrict__ t,
                                            int half) {
    f8 acc = {};
#if __has_builtin(__builtin_amdgcn_wmma_f32_16x16x4_f32)
    for (int k0 = 0; k0 < HN_; k0 += 4) {
        int kk = k0 + 2 * half;
        f2 av = *(const f2*)(Xrow + kk);
        f2 sv = *(const f2*)(s + kk);
        f2 tv = *(const f2*)(t + kk);
        f2 a;  a[0] = av[0] * sv[0] + tv[0];  a[1] = av[1] * sv[1] + tv[1];
        f2 b  = *(const f2*)(Wrow + kk);
        acc = __builtin_amdgcn_wmma_f32_16x16x4_f32(false, a, false, b,
                                                    (short)0, acc, false, false);
    }
#else
    for (int k = 0; k < HN_; ++k) acc[0] += (Xrow[k] * s[k] + t[k]) * Wrow[k];
#endif
    return acc;
}

// ---------------------------------------------------------------------------
// Kernel 3: zp = bn(z) @ cls_W^T + cls_b    (4096 x 768), f32 WMMA.
// ---------------------------------------------------------------------------
__global__ void k_zp(const float* __restrict__ z,
                     const float* __restrict__ sz, const float* __restrict__ tz,
                     const float* __restrict__ W,  const float* __restrict__ bias,
                     float* __restrict__ zp) {
    const int lane = threadIdx.x & 31;
    const int wid  = blockIdx.x * 4 + (threadIdx.x >> 5);
    const int NT   = HN_ / 16;                   // 48
    const int m0   = (wid / NT) * 16, n0 = (wid % NT) * 16;
    const int half = lane >> 4, nl = lane & 15;

    const float* Xrow = z + (size_t)(m0 + nl) * HN_;
    const float* Wrow = W + (size_t)(n0 + nl) * HN_;
    f8 acc = gemm_f32_tile(Xrow, Wrow, sz, tz, half);

    int n = n0 + nl;
    float bb = bias[n];
#pragma unroll
    for (int v = 0; v < 8; ++v)
        zp[(size_t)(m0 + v + 8 * half) * HN_ + n] = acc[v] + bb;
}

// ---------------------------------------------------------------------------
// Kernel 4: tp = bn(theta) @ label_W^T + label_b fused with
// logits[jl] += sum_n tp[jl,n] * zp[jl/5, n].  tp is never materialized.
// ---------------------------------------------------------------------------
__global__ void k_logits(const float* __restrict__ th,
                         const float* __restrict__ st, const float* __restrict__ tt,
                         const float* __restrict__ W,  const float* __restrict__ bias,
                         const float* __restrict__ zp, float* __restrict__ out) {
    const int lane = threadIdx.x & 31;
    const int wid  = blockIdx.x * 4 + (threadIdx.x >> 5);
    const int NT   = HN_ / 16;                   // 48
    const int m0   = (wid / NT) * 16, n0 = (wid % NT) * 16;
    const int half = lane >> 4, nl = lane & 15;

    const float* Xrow = th + (size_t)(m0 + nl) * HN_;
    const float* Wrow = W  + (size_t)(n0 + nl) * HN_;
    f8 acc = gemm_f32_tile(Xrow, Wrow, st, tt, half);

    int n = n0 + nl;
    float lb = bias[n];
#pragma unroll
    for (int v = 0; v < 8; ++v) {
        int jl = m0 + v + 8 * half;
        int j  = jl / 5;
        float contrib = (acc[v] + lb) * zp[(size_t)j * HN_ + n];
#pragma unroll
        for (int m = 1; m < 16; m <<= 1) contrib += __shfl_xor(contrib, m, 32);
        if (nl == 0) atomicAdd(&out[jl], contrib);
    }
}

// ---------------------------------------------------------------------------
// async global -> LDS copy (ASYNCcnt-tracked), 16 bytes per op.  GV mode:
// vdst = LDS byte offset (low 32 bits of the flat shared address), vaddr =
// 64-bit global address.
// ---------------------------------------------------------------------------
__device__ __forceinline__ void async_copy_b128(uint32_t lds_off, const void* gaddr) {
    asm volatile("global_load_async_to_lds_b128 %0, %1, off"
                 :: "v"(lds_off), "v"(gaddr) : "memory");
}

// ---------------------------------------------------------------------------
// Kernel 5: big fused contrastive GEMM, LDS-staged + double-buffered async.
// S[i, jl] = <z_norm[i], theta_norm[jl]>  (4096 x 20480, K=768), f16 WMMA.
// Block = 512 threads (16 waves) owning a 128x128 tile; each wave computes a
// 32x32 quadrant (2x2 WMMA tiles) from shared LDS tiles (64 FLOP/byte vs L2).
// Epilogue fuses exp(S/T) + masked row sums (pos_z/neg_z) and column sums
// (pos_t/neg_t) via the identity expT[p,q] = expS[q,p,labels[p]].
// ---------------------------------------------------------------------------
__global__ void __launch_bounds__(512)
k_contrast(const _Float16* __restrict__ znh, const _Float16* __restrict__ thh,
           const int* __restrict__ lab,
           float* __restrict__ posz, float* __restrict__ negz,
           float* __restrict__ post, float* __restrict__ negt) {
    __shared__ __attribute__((aligned(16))) _Float16 Ab[2][128 * 32];   // 8 KB x2
    __shared__ __attribute__((aligned(16))) _Float16 Bb[2][128 * 32];   // 8 KB x2

    const int tid  = threadIdx.x;
    const int lane = tid & 31;
    const int w    = tid >> 5;                   // 0..15
    const int half = lane >> 4, nl = lane & 15;

    const int NB  = SMN_ / 128;                  // 160
    const int m0  = (blockIdx.x / NB) * 128;
    const int n0  = (blockIdx.x % NB) * 128;
    const int r0  = 32 * (w >> 2);               // wave quadrant inside block
    const int c0  = 32 * (w & 3);
    const int m0w = m0 + r0, n0w = n0 + c0;

    // copy-role for this thread: one 16B chunk of A and one of B per K-step
    const int crow = tid >> 2, cpart = tid & 3;  // row 0..127, 16B part 0..3
    const _Float16* gA = znh + (size_t)(m0 + crow) * HN_ + cpart * 8;
    const _Float16* gB = thh + (size_t)(n0 + crow) * HN_ + cpart * 8;
    const uint32_t  lA0 = (uint32_t)(uintptr_t)&Ab[0][crow * 32 + cpart * 8];
    const uint32_t  lA1 = (uint32_t)(uintptr_t)&Ab[1][crow * 32 + cpart * 8];
    const uint32_t  lB0 = (uint32_t)(uintptr_t)&Bb[0][crow * 32 + cpart * 8];
    const uint32_t  lB1 = (uint32_t)(uintptr_t)&Bb[1][crow * 32 + cpart * 8];

    f8 acc[2][2] = {};
    const int NSTEP = HN_ / 32;                  // 24

    // prologue: fill buffer 0
    async_copy_b128(lA0, gA);
    async_copy_b128(lB0, gB);

    for (int s2 = 0; s2 < NSTEP; ++s2) {
        const int cur = s2 & 1;
        if (s2 + 1 < NSTEP) {
            const _Float16* ga = gA + (size_t)(s2 + 1) * 32;
            const _Float16* gb = gB + (size_t)(s2 + 1) * 32;
            async_copy_b128(cur ? lA0 : lA1, ga);
            async_copy_b128(cur ? lB0 : lB1, gb);
            asm volatile("s_wait_asynccnt 0x2" ::: "memory");   // current buffer landed
        } else {
            asm volatile("s_wait_asynccnt 0x0" ::: "memory");
        }
        __syncthreads();

        // compute on buffer `cur`
        h16 afr[2], bfr[2];
#pragma unroll
        for (int tm = 0; tm < 2; ++tm) {
            const _Float16* arow = &Ab[cur][(r0 + 16 * tm + nl) * 32];
            h8 lo = *(const h8*)(arow + 8 * half);
            h8 hi = *(const h8*)(arow + 16 + 8 * half);
#pragma unroll
            for (int t = 0; t < 8; ++t) { afr[tm][t] = lo[t]; afr[tm][8 + t] = hi[t]; }
        }
#pragma unroll
        for (int tn = 0; tn < 2; ++tn) {
            const _Float16* brow = &Bb[cur][(c0 + 16 * tn + nl) * 32];
            h8 lo = *(const h8*)(brow + 8 * half);
            h8 hi = *(const h8*)(brow + 16 + 8 * half);
#pragma unroll
            for (int t = 0; t < 8; ++t) { bfr[tn][t] = lo[t]; bfr[tn][8 + t] = hi[t]; }
        }
        acc[0][0] = __builtin_amdgcn_wmma_f32_16x16x32_f16(false, afr[0], false, bfr[0], (short)0, acc[0][0], false, false);
        acc[0][1] = __builtin_amdgcn_wmma_f32_16x16x32_f16(false, afr[0], false, bfr[1], (short)0, acc[0][1], false, false);
        acc[1][0] = __builtin_amdgcn_wmma_f32_16x16x32_f16(false, afr[1], false, bfr[0], (short)0, acc[1][0], false, false);
        acc[1][1] = __builtin_amdgcn_wmma_f32_16x16x32_f16(false, afr[1], false, bfr[1], (short)0, acc[1][1], false, false);

        __syncthreads();   // protect buffer reuse (re-issued next iteration)
    }

    // ---- fused epilogue ----
    int jcol[2], lcol[2], lj[2];
#pragma unroll
    for (int tn = 0; tn < 2; ++tn) {
        int n = n0w + 16 * tn + nl;
        jcol[tn] = n / 5;
        lcol[tn] = n - 5 * jcol[tn];
        lj[tn]   = lab[jcol[tn]];
    }
    float cp[2] = {0.f, 0.f}, cn[2] = {0.f, 0.f};

#pragma unroll
    for (int tm = 0; tm < 2; ++tm) {
        const int ibase = m0w + 16 * tm + 8 * half;
        int li[8];
#pragma unroll
        for (int v = 0; v < 8; ++v) li[v] = lab[ibase + v];

        float rp[8] = {0.f,0.f,0.f,0.f,0.f,0.f,0.f,0.f};
        float rn[8] = {0.f,0.f,0.f,0.f,0.f,0.f,0.f,0.f};
#pragma unroll
        for (int tn = 0; tn < 2; ++tn) {
#pragma unroll
            for (int v = 0; v < 8; ++v) {
                float e = __expf(acc[tm][tn][v] * INV_T);
                int  i    = ibase + v;
                bool same = (li[v] == lj[tn]);
                bool self = (i == jcol[tn]);
                bool atli = (lcol[tn] == li[v]);
                bool atlj = (lcol[tn] == lj[tn]);
                rn[v]  += same                    ? 0.f : e;     // neg_z (all l)
                rp[v]  += (same && !self && atli) ? e : 0.f;     // pos_z
                cp[tn] += (same && !self && atlj) ? e : 0.f;     // pos_t
                cn[tn] += (!same && atlj)         ? e : 0.f;     // neg_t
            }
        }
#pragma unroll
        for (int v = 0; v < 8; ++v) {
            float a = rp[v], b = rn[v];
#pragma unroll
            for (int m = 1; m < 16; m <<= 1) { a += __shfl_xor(a, m, 32); b += __shfl_xor(b, m, 32); }
            if (nl == 0) {
                atomicAdd(&posz[ibase + v], a);
                atomicAdd(&negz[ibase + v], b);
            }
        }
    }
#pragma unroll
    for (int tn = 0; tn < 2; ++tn) {
        cp[tn] += __shfl_xor(cp[tn], 16, 32);
        cn[tn] += __shfl_xor(cn[tn], 16, 32);
        if (half == 0) {
            atomicAdd(&post[jcol[tn]], cp[tn]);
            atomicAdd(&negt[jcol[tn]], cn[tn]);
        }
    }
}

// ---------------------------------------------------------------------------
// Kernel 6: final loss reduction.
// ---------------------------------------------------------------------------
__global__ void k_loss(const float* __restrict__ posz, const float* __restrict__ negz,
                       const float* __restrict__ post, const float* __restrict__ negt,
                       const int* __restrict__ lab, float* __restrict__ out) {
    __shared__ int   hist[LN_];
    __shared__ float red[256];
    int tid = threadIdx.x;
    if (tid < LN_) hist[tid] = 0;
    __syncthreads();
    for (int i = tid; i < BN_; i += 256) atomicAdd(&hist[lab[i]], 1);
    __syncthreads();

    float acc = 0.0f;
    for (int i = tid; i < BN_; i += 256) {
        if (hist[lab[i]] > 1) {
            float pz = posz[i], nz = negz[i];
            acc += -logf(pz / (pz + nz + 1e-8f));
            float pt = post[i], nt = negt[i];
            acc += -logf(pt / (pt + nt + 1e-8f));
        }
    }
    red[tid] = acc;
    __syncthreads();
    for (int s = 128; s > 0; s >>= 1) {
        if (tid < s) red[tid] += red[tid + s];
        __syncthreads();
    }
    if (tid == 0) out[SMN_] = red[0] / (float)BN_;
}

// ---------------------------------------------------------------------------
extern "C" void kernel_launch(void* const* d_in, const int* in_sizes, int n_in,
                              void* d_out, int out_size, void* d_ws, size_t ws_size,
                              hipStream_t stream) {
    const float* seq   = (const float*)d_in[0];
    const float* cg    = (const float*)d_in[1];
    const float* cb    = (const float*)d_in[2];
    const float* cW    = (const float*)d_in[3];
    const float* cbias = (const float*)d_in[4];
    const float* lg    = (const float*)d_in[5];
    const float* lbeta = (const float*)d_in[6];
    const float* lW    = (const float*)d_in[7];
    const float* lbias = (const float*)d_in[8];
    const int*   lab   = (const int*)d_in[9];
    float* out = (float*)d_out;
    float* ws  = (float*)d_ws;

    // workspace layout (floats); f16 regions 32B-aligned by construction
    float* z    = ws;                              // 4096*768
    float* th   = z  + (size_t)BN_ * HN_;          // 20480*768
    float* zp   = th + (size_t)SMN_ * HN_;         // 4096*768
    float* sz   = zp + (size_t)BN_ * HN_;          // 768
    float* tz   = sz + HN_;
    float* st   = tz + HN_;
    float* tt   = st + HN_;
    float* posz = tt + HN_;                        // 4*4096 contiguous
    float* negz = posz + BN_;
    float* post = negz + BN_;
    float* negt = post + BN_;
    _Float16* znh = (_Float16*)(negt + BN_);       // 4096*768 halfs
    _Float16* thh = znh + (size_t)BN_ * HN_;       // 20480*768 halfs

    k_init    <<<(4 * BN_ + SMN_ + 1 + 255) / 256, 256, 0, stream>>>(posz, out);
    k_extract <<<BN_, 192, 0, stream>>>(seq, z, th, znh, thh);
    k_colstats<<<6, 256, 0, stream>>>(z, th, cg, cb, lg, lbeta, sz, tz, st, tt);
    k_zp      <<<(BN_ / 16) * (HN_ / 16) / 4, 128, 0, stream>>>(z, sz, tz, cW, cbias, zp);
    k_logits  <<<(SMN_ / 16) * (HN_ / 16) / 4, 128, 0, stream>>>(th, st, tt, lW, lbias, zp, out);
    k_contrast<<<(BN_ / 128) * (SMN_ / 128), 512, 0, stream>>>(znh, thh, lab, posz, negz, post, negt);
    k_loss    <<<1, 256, 0, stream>>>(posz, negz, post, negt, lab, out);
}
